// Expert_62036507623570
// MI455X (gfx1250) — matrix operand order repaired and checked by reading, not compiled, not run on hardware
//
#include <hip/hip_runtime.h>

typedef __attribute__((ext_vector_type(16))) _Float16 v16h;
typedef __attribute__((ext_vector_type(8)))  _Float16 v8h;
typedef __attribute__((ext_vector_type(8)))  float    v8f;

#define N_NODES 50000
#define DEG     16
#define N_EDGES (N_NODES*DEG)
#define DIM     128
#define HID     128
#define W1H_ELEMS (8*8*32*16)   // [jtile][kchunk][lane][16 halves] = 32768 halves (64 KB)
#define W2H_ELEMS (8*4*32*16)   // 16384 halves (32 KB)

// ---------------- pre-kernels: f32 -> f16 conversion / fragment repack ----------------

__global__ void cvt_feat_kernel(const float* __restrict__ f, _Float16* __restrict__ o, int n) {
  for (int i = blockIdx.x*blockDim.x + threadIdx.x; i < n; i += gridDim.x*blockDim.x)
    o[i] = (_Float16)f[i];
}

// B-fragment packing used here (contiguous K, same order as A; not AMD's table order): lanes 0-15 hold N=lane, K=0..15 in halves 0..15;
// lanes 16-31 hold N=lane-16, K=16..31.
__global__ void pack_w1_kernel(const float* __restrict__ w1, _Float16* __restrict__ o) {
  int t = blockIdx.x*blockDim.x + threadIdx.x;
  if (t >= W1H_ELEMS) return;
  int h    = t & 15;
  int lane = (t >> 4) & 31;
  int ch   = (t >> 9) & 7;
  int jt   = t >> 12;
  int K = ch*32 + ((lane >= 16) ? 16 : 0) + h;          // K < 256 (row 256 handled separately)
  int j = jt*16 + (lane & 15);
  o[t] = (_Float16)w1[K*HID + j];
}

__global__ void pack_w2_kernel(const float* __restrict__ w2, _Float16* __restrict__ o) {
  int t = blockIdx.x*blockDim.x + threadIdx.x;
  if (t >= W2H_ELEMS) return;
  int h    = t & 15;
  int lane = (t >> 4) & 31;
  int ch   = (t >> 9) & 3;
  int jt   = t >> 11;
  int K = ch*32 + ((lane >= 16) ? 16 : 0) + h;
  int j = jt*16 + (lane & 15);
  o[t] = (_Float16)w2[K*HID + j];
}

// ---------------- main fused kernel: one wave32 per node ----------------

__device__ __forceinline__ v16h make_frag(const _Float16* lo, const _Float16* hi) {
  v8h a = *(const v8h*)lo;
  v8h b = *(const v8h*)hi;
  v16h r;
  #pragma unroll
  for (int i = 0; i < 8; ++i) { r[i] = a[i]; r[i+8] = b[i]; }
  return r;
}

__global__ __launch_bounds__(256) void edge_score_kernel(
    const int* __restrict__ indices, const float* __restrict__ values,
    const float* __restrict__ temperature,
    const float* __restrict__ w1, const float* __restrict__ b1,
    const float* __restrict__ b2, const float* __restrict__ w3,
    const float* __restrict__ b3,
    const _Float16* __restrict__ featH,
    const _Float16* __restrict__ w1h, const _Float16* __restrict__ w2h,
    float* __restrict__ out)
{
  // WGP has 320 KB LDS: stage shared weight fragments once per block (reused by 8 waves x 1 node each)
  __shared__ _Float16 w1lds[W1H_ELEMS];       // 64 KB
  __shared__ _Float16 w2lds[W2H_ELEMS];       // 32 KB
  __shared__ _Float16 h1buf[8][16*136];       // 34.8 KB, padded stride spreads banks
  __shared__ float    zbuf[8][16];

  const int  lane  = threadIdx.x & 31;
  const int  wid   = threadIdx.x >> 5;
  const int  node  = blockIdx.x * 8 + wid;    // 6250 blocks * 8 waves == 50000 nodes
  const int  m     = lane & 15;               // edge-in-node / N-col within tile
  const bool hi    = lane >= 16;
  const int  kb    = hi ? 16 : 0;             // A-fragment K-half offset
  const int  mbase = hi ? 8 : 0;              // C-fragment row offset
  const int  ebase = node * DEG;

  const int   col  = indices[N_EDGES + ebase + m];   // indices[1][e]
  const float vloc = values[ebase + m];              // VDIM == 1
  const _Float16* rowp = featH + (size_t)node * DIM;
  const _Float16* colp = featH + (size_t)col  * DIM;
  __builtin_prefetch((const void*)colp, 0, 3);       // global_prefetch: hide random-gather latency
  _Float16* h1p = &h1buf[wid][0];

  // ---- cooperative staging of packed weights into LDS (contiguous b128 copies) ----
  {
    const v8h* s1 = (const v8h*)w1h;  v8h* d1 = (v8h*)w1lds;
    for (int i = threadIdx.x; i < W1H_ELEMS/8; i += 256) d1[i] = s1[i];
    const v8h* s2 = (const v8h*)w2h;  v8h* d2 = (v8h*)w2lds;
    for (int i = threadIdx.x; i < W2H_ELEMS/8; i += 256) d2[i] = s2[i];
  }
  __syncthreads();

  float zacc[8];
  #pragma unroll
  for (int V = 0; V < 8; ++V) zacc[V] = 0.f;

  // ---------------- layer 1: [16,256] x [256,128] (+ rank-1 val term folded into C init) ----
  for (int jt = 0; jt < 8; ++jt) {
    const int   jj   = jt*16 + m;
    const float b1j  = b1[jj];
    const float w257 = w1[256*HID + jj];             // last input row (edge value weight)
    v8f c;
    #pragma unroll
    for (int V = 0; V < 8; ++V) {
      float vm = __shfl(vloc, mbase + V, 16);        // broadcast values[edge row mbase+V]
      c[V] = b1j + vm * w257;
    }
    #pragma unroll
    for (int ch = 0; ch < 8; ++ch) {
      // chunks 0-3: broadcast row-node features; 4-7: gathered col-node features
      const _Float16* src = (ch < 4 ? rowp + ch*32 : colp + (ch-4)*32) + kb;
      v16h a = make_frag(src, src + 8);              // A 16x32 f16 layout (K contiguous 0-15/16-31)
      const _Float16* bp = w1lds + ((jt*8 + ch)*32 + lane) * 16;
      v16h b = make_frag(bp, bp + 8);
      c = __builtin_amdgcn_wmma_f32_16x16x32_f16(false, a, false, b, (short)0, c, false, false);
    }
    #pragma unroll
    for (int V = 0; V < 8; ++V) {                    // relu -> f16 -> LDS (transpose staging)
      float r = c[V] > 0.f ? c[V] : 0.f;
      h1p[(mbase + V)*136 + jj] = (_Float16)r;
    }
  }
  asm volatile("s_wait_dscnt 0" ::: "memory");       // CDNA5 split DS counter (wave-local h1)

  // ---------------- layer 2: [16,128] x [128,128]; layer 3 folded into epilogue ------------
  for (int jt = 0; jt < 8; ++jt) {
    const int   jj  = jt*16 + m;
    const float b2j = b2[jj];
    v8f c;
    #pragma unroll
    for (int V = 0; V < 8; ++V) c[V] = b2j;
    #pragma unroll
    for (int ch = 0; ch < 4; ++ch) {
      const _Float16* src = h1p + m*136 + ch*32 + kb;
      v16h a = make_frag(src, src + 8);
      const _Float16* bp = w2lds + ((jt*4 + ch)*32 + lane) * 16;
      v16h b = make_frag(bp, bp + 8);
      c = __builtin_amdgcn_wmma_f32_16x16x32_f16(false, a, false, b, (short)0, c, false, false);
    }
    const float w3j = w3[jj];
    #pragma unroll
    for (int V = 0; V < 8; ++V) {                    // z[m] partial: relu(h2[m,jj]) * w3[jj]
      float r = c[V] > 0.f ? c[V] : 0.f;
      zacc[V] += r * w3j;
    }
  }

  // reduce z over the 16 lanes of each half (rows 0-7 in lanes 0-15, rows 8-15 in 16-31)
  #pragma unroll
  for (int mask = 1; mask < 16; mask <<= 1) {
    #pragma unroll
    for (int V = 0; V < 8; ++V) zacc[V] += __shfl_xor(zacc[V], mask, 16);
  }
  if (m == 0) {                                      // lane 0 -> z[0..7], lane 16 -> z[8..15]
    #pragma unroll
    for (int V = 0; V < 8; ++V) zbuf[wid][mbase + V] = zacc[V];
  }
  asm volatile("s_wait_dscnt 0" ::: "memory");

  // ---------------- per-node scoring, all wave-local (m = lane & 15) -----------------------
  float z = zbuf[wid][m] + b3[0];
  const float invT = 1.0f / temperature[0];

  float mx = z;                                      // segment softmax #1
  #pragma unroll
  for (int mask = 1; mask < 16; mask <<= 1) mx = fmaxf(mx, __shfl_xor(mx, mask, 16));
  float e = expf(z - mx);
  float s = e;
  #pragma unroll
  for (int mask = 1; mask < 16; mask <<= 1) s += __shfl_xor(s, mask, 16);
  float pi = e / s;

  float hard = 1.0f / (1.0f + expf(-(logf(pi + 1e-8f) * invT)));  // hard-concrete (eval path)
  hard = fminf(fmaxf(hard, 0.0f), 1.0f);

  float mx2 = hard;                                  // segment softmax #2
  #pragma unroll
  for (int mask = 1; mask < 16; mask <<= 1) mx2 = fmaxf(mx2, __shfl_xor(mx2, mask, 16));
  float e2 = expf(hard - mx2);
  float s2 = e2;
  #pragma unroll
  for (int mask = 1; mask < 16; mask <<= 1) s2 += __shfl_xor(s2, mask, 16);
  float y = e2 / s2;

  // stable descending rank (matches lexsort tie order); threshold = 8th largest (k = 16*0.5)
  int rank = 0;
  #pragma unroll
  for (int jl = 0; jl < 16; ++jl) {
    float yj = __shfl(y, jl, 16);
    rank += ((yj > y) || (yj == y && jl < m)) ? 1 : 0;
  }
  float cand = (rank == 7) ? y : -3.0e38f;
  float thre = cand;
  #pragma unroll
  for (int mask = 1; mask < 16; mask <<= 1) thre = fmaxf(thre, __shfl_xor(thre, mask, 16));

  float o = (y - thre + 1e-7f > 0.0f) ? y : 0.0f;
  if (!hi) out[ebase + m] = o;
}

// ---------------- host entry ----------------

extern "C" void kernel_launch(void* const* d_in, const int* in_sizes, int n_in,
                              void* d_out, int out_size, void* d_ws, size_t ws_size,
                              hipStream_t stream) {
  (void)in_sizes; (void)n_in; (void)out_size; (void)ws_size;
  const float* features    = (const float*)d_in[0];
  const int*   indices     = (const int*)  d_in[1];
  const float* values      = (const float*)d_in[2];
  const float* temperature = (const float*)d_in[3];
  const float* w1 = (const float*)d_in[4];
  const float* b1 = (const float*)d_in[5];
  const float* w2 = (const float*)d_in[6];
  const float* b2 = (const float*)d_in[7];
  const float* w3 = (const float*)d_in[8];
  const float* b3 = (const float*)d_in[9];
  float* out = (float*)d_out;

  // workspace: featH (12.8 MB) + packed f16 weights (~96 KB)
  _Float16* featH = (_Float16*)d_ws;
  _Float16* w1h   = featH + (size_t)N_NODES * DIM;
  _Float16* w2h   = w1h + W1H_ELEMS;

  const int nfeat = N_NODES * DIM;
  cvt_feat_kernel<<<(nfeat + 255)/256, 256, 0, stream>>>(features, featH, nfeat);
  pack_w1_kernel<<<(W1H_ELEMS + 255)/256, 256, 0, stream>>>(w1, w1h);
  pack_w2_kernel<<<(W2H_ELEMS + 255)/256, 256, 0, stream>>>(w2, w2h);

  edge_score_kernel<<<N_NODES/8, 256, 0, stream>>>(
      indices, values, temperature, w1, b1, b2, w3, b3, featH, w1h, w2h, out);
}
